// BinaryLinearED_36412732736203
// MI455X (gfx1250) — matrix operand order repaired and checked by reading, not compiled, not run on hardware
//
#include <hip/hip_runtime.h>

typedef _Float16 v8h  __attribute__((ext_vector_type(8)));
typedef _Float16 v16h __attribute__((ext_vector_type(16)));
typedef float    v8f  __attribute__((ext_vector_type(8)));

#define M_DIM 8192
#define N_DIM 4096
#define K_DIM 4096

// ---------------- conversion kernels ----------------

__global__ __launch_bounds__(256) void cvt_x_f16(const float* __restrict__ x,
                                                 _Float16* __restrict__ y) {
    size_t i = ((size_t)blockIdx.x * blockDim.x + threadIdx.x) * 8;
    float4 a = *(const float4*)(x + i);
    float4 b = *(const float4*)(x + i + 4);
    v8h o;
    o[0] = (_Float16)a.x; o[1] = (_Float16)a.y; o[2] = (_Float16)a.z; o[3] = (_Float16)a.w;
    o[4] = (_Float16)b.x; o[5] = (_Float16)b.y; o[6] = (_Float16)b.z; o[7] = (_Float16)b.w;
    *(v8h*)(y + i) = o;
}

__device__ __forceinline__ _Float16 signh(float v) {
    return (_Float16)((v > 0.0f) ? 1.0f : ((v < 0.0f) ? -1.0f : 0.0f));
}

__global__ __launch_bounds__(256) void cvt_w_sign_f16(const float* __restrict__ w,
                                                      _Float16* __restrict__ y) {
    size_t i = ((size_t)blockIdx.x * blockDim.x + threadIdx.x) * 8;
    float4 a = *(const float4*)(w + i);
    float4 b = *(const float4*)(w + i + 4);
    v8h o;
    o[0] = signh(a.x); o[1] = signh(a.y); o[2] = signh(a.z); o[3] = signh(a.w);
    o[4] = signh(b.x); o[5] = signh(b.y); o[6] = signh(b.z); o[7] = signh(b.w);
    *(v8h*)(y + i) = o;
}

// ---------------- WMMA GEMM: C[M,N] = A[M,K] * B[N,K]^T + bias ----------------

constexpr int BM = 128;
constexpr int BN = 128;
constexpr int BK = 64;
constexpr int LDT = BK + 8;                 // padded LDS row stride (halves)
constexpr int NSTAGE = K_DIM / BK;          // 64
constexpr unsigned BUF_BYTES = BM * LDT * 2;  // 18432 B per buffer (same for A and B)

__device__ __forceinline__ void wait_async0() {
#if __has_builtin(__builtin_amdgcn_s_wait_asynccnt)
    __builtin_amdgcn_s_wait_asynccnt(0);
#else
    asm volatile("s_wait_asynccnt 0x0" ::: "memory");
#endif
}

__global__ __launch_bounds__(256) void bin_gemm_f16(const _Float16* __restrict__ A,
                                                    const _Float16* __restrict__ B,
                                                    const float* __restrict__ bias,
                                                    float* __restrict__ C) {
    __shared__ _Float16 sA[2][BM * LDT];
    __shared__ _Float16 sB[2][BN * LDT];

    const int t     = threadIdx.x;
    const int lane  = t & 31;
    const int wave  = t >> 5;
    const int waveM = wave >> 2;            // 0..1 -> 64 rows each
    const int waveN = wave & 3;             // 0..3 -> 32 cols each
    const int mBlk  = blockIdx.y * BM;
    const int nBlk  = blockIdx.x * BN;

    // async-copy mapping: 1024 16B chunks per tile, 4 per thread (i*32 rows apart)
    const int crow = t >> 3;                // 0..31
    const int ccol = (t & 7) * 8;           // halves
    unsigned voffA = (unsigned)((mBlk + crow) * K_DIM + ccol) * 2u;   // bytes
    unsigned voffB = (unsigned)((nBlk + crow) * K_DIM + ccol) * 2u;   // bytes
    const unsigned ldsA0 = (unsigned)(size_t)&sA[0][crow * LDT + ccol];
    const unsigned ldsB0 = (unsigned)(size_t)&sB[0][crow * LDT + ccol];

    auto issue_stage = [&](unsigned bufoff) {
#pragma unroll
        for (int i = 0; i < 4; ++i) {
            unsigned la = ldsA0 + bufoff + (unsigned)(i * 32 * LDT) * 2u;
            unsigned ga = voffA + (unsigned)(i * 32 * K_DIM) * 2u;
            asm volatile("global_load_async_to_lds_b128 %0, %1, %2"
                         :: "v"(la), "v"(ga), "s"(A) : "memory");
            unsigned lb = ldsB0 + bufoff + (unsigned)(i * 32 * LDT) * 2u;
            unsigned gb = voffB + (unsigned)(i * 32 * K_DIM) * 2u;
            asm volatile("global_load_async_to_lds_b128 %0, %1, %2"
                         :: "v"(lb), "v"(gb), "s"(B) : "memory");
        }
    };

    v8f acc[4][2] = {};

    // fragment lane mapping (ISA 7.12.2)
    const int frow = lane & 15;
    const int kA   = (lane >> 4) * 8;       // A: interleaved 8-half K runs
    const int kB   = (lane >> 4) * 8;       // B: interleaved 8-half K runs

    // prologue: DMA stage 0 into buffer 0
    issue_stage(0);
    voffA += BK * 2;  voffB += BK * 2;
    wait_async0();
    __syncthreads();

    for (int s = 0; s < NSTAGE; ++s) {
        const int cur = s & 1;
        if (s + 1 < NSTAGE) {
            issue_stage(((s + 1) & 1) ? BUF_BYTES : 0u);
            voffA += BK * 2;  voffB += BK * 2;
        }

        const _Float16* a0 = &sA[cur][0];
        const _Float16* b0 = &sB[cur][0];

#pragma unroll
        for (int kk = 0; kk < BK; kk += 32) {
            v16h af[4], bf[2];
#pragma unroll
            for (int mt = 0; mt < 4; ++mt) {
                const _Float16* p = a0 + (waveM * 64 + mt * 16 + frow) * LDT + kk + kA;
                union { v16h v; v8h h[2]; } u;
                u.h[0] = *(const v8h*)(p);
                u.h[1] = *(const v8h*)(p + 16);
                af[mt] = u.v;
            }
#pragma unroll
            for (int nt = 0; nt < 2; ++nt) {
                const _Float16* p = b0 + (waveN * 32 + nt * 16 + frow) * LDT + kk + kB;
                union { v16h v; v8h h[2]; } u;
                u.h[0] = *(const v8h*)(p);
                u.h[1] = *(const v8h*)(p + 16);
                bf[nt] = u.v;
            }
#pragma unroll
            for (int mt = 0; mt < 4; ++mt)
#pragma unroll
                for (int nt = 0; nt < 2; ++nt)
                    acc[mt][nt] = __builtin_amdgcn_wmma_f32_16x16x32_f16(
                        false, af[mt], false, bf[nt], (short)0, acc[mt][nt], false, false);
        }

        if (s + 1 < NSTAGE) {
            wait_async0();        // my DMAs for next stage have landed
            __syncthreads();      // everyone's DMAs visible, everyone done reading cur
        }
    }

    // epilogue: C/D layout -> lane = N col (lane&15), M row = (lane>>4)*8 + vgpr
    const int cN = lane & 15;
    const int cM = (lane >> 4) * 8;
#pragma unroll
    for (int mt = 0; mt < 4; ++mt) {
#pragma unroll
        for (int nt = 0; nt < 2; ++nt) {
            int col  = nBlk + waveN * 32 + nt * 16 + cN;
            float bv = bias[col];
#pragma unroll
            for (int r = 0; r < 8; ++r) {
                int row = mBlk + waveM * 64 + mt * 16 + cM + r;
                C[(size_t)row * N_DIM + col] = acc[mt][nt][r] + bv;
            }
        }
    }
}

// ---------------- launch ----------------

extern "C" void kernel_launch(void* const* d_in, const int* in_sizes, int n_in,
                              void* d_out, int out_size, void* d_ws, size_t ws_size,
                              hipStream_t stream) {
    const float* x    = (const float*)d_in[0];
    const float* w    = (const float*)d_in[1];
    const float* bias = (const float*)d_in[2];
    float* out        = (float*)d_out;

    _Float16* xh = (_Float16*)d_ws;                     // 8192*4096*2 = 64 MB
    _Float16* wh = xh + (size_t)M_DIM * K_DIM;          // 4096*4096*2 = 32 MB

    cvt_x_f16<<<(M_DIM * (size_t)K_DIM / 8) / 256, 256, 0, stream>>>(x, xh);
    cvt_w_sign_f16<<<(N_DIM * (size_t)K_DIM / 8) / 256, 256, 0, stream>>>(w, wh);

    dim3 grid(N_DIM / BN, M_DIM / BM);   // 32 x 64 blocks
    bin_gemm_f16<<<grid, 256, 0, stream>>>(xh, wh, bias, out);
}
